// LitePT_39092792328864
// MI455X (gfx1250) — matrix operand order repaired and checked by reading, not compiled, not run on hardware
//
#include <hip/hip_runtime.h>

// ---------------------------------------------------------------------------
// Problem constants
// ---------------------------------------------------------------------------
#define NPTS   32768
#define CCH    768
#define HEADS  8
#define HD     96          // head dim
#define WIN    256         // window size K
#define PWIN   (NPTS/WIN)  // 128 windows
#define C3     (3*CCH)     // 2304

typedef __bf16 bf16;
typedef __attribute__((ext_vector_type(8)))  __bf16 bf16x8;
typedef __attribute__((ext_vector_type(16))) __bf16 bf16x16;
typedef __attribute__((ext_vector_type(8)))  float  v8f;

// ---------------------------------------------------------------------------
// Fragment helpers (CDNA5 WMMA 16x16x32 bf16, wave32 layouts per ISA 7.12.2)
//   A-frag, lane L: m = L&15, sel = L>>4; two contiguous 8-elem runs at
//                   k = k0 + sel*8  and  k = k0 + 16 + sel*8
//   B-frag (from pre-transposed operand Bt[N][K]), lane L: n = L&15,
//                   two contiguous 8-elem runs at k = k0 + sel*8 and k0 + 16 + sel*8 (as A)
//   C/D, lane L: col = L&15, rows = sel*8 + r (r = VGPR index 0..7)
// ---------------------------------------------------------------------------
__device__ __forceinline__ bf16x16 frag_ld(const bf16* p0, const bf16* p1) {
  union { bf16x16 v; bf16x8 h[2]; } u;
  u.h[0] = *(const bf16x8*)p0;
  u.h[1] = *(const bf16x8*)p1;
  return u.v;
}

__device__ __forceinline__ v8f wmma_bf16(bf16x16 a, bf16x16 b, v8f c) {
  return __builtin_amdgcn_wmma_f32_16x16x32_bf16(
      /*neg_a=*/false, a, /*neg_b=*/false, b,
      /*c_mod=*/(short)0, c, /*reuse_a=*/false, /*reuse_b=*/false);
}

// Async global->LDS copy of 16 bytes per lane (gfx1250, ASYNCcnt-tracked).
__device__ __forceinline__ void async_ld_b128(unsigned lds_off, unsigned goff,
                                              const void* base) {
  asm volatile("global_load_async_to_lds_b128 %0, %1, %2"
               :: "v"(lds_off), "v"(goff), "s"(base) : "memory");
}
// ASYNCcnt decrements in issue order -> waiting to <=12 outstanding
// guarantees the first 12 issued loads (the K tile) have landed.
__device__ __forceinline__ void wait_async12() {
  asm volatile("s_wait_asynccnt 0xc" ::: "memory");
}
__device__ __forceinline__ void wait_async0() {
  asm volatile("s_wait_asynccnt 0x0" ::: "memory");
}

// ---------------------------------------------------------------------------
// fp32 -> bf16 convert (flat)
// ---------------------------------------------------------------------------
__global__ __launch_bounds__(256) void cvt_bf16_kernel(
    const float* __restrict__ src, bf16* __restrict__ dst, int n) {
  for (int i = blockIdx.x * blockDim.x + threadIdx.x; i < n;
       i += gridDim.x * blockDim.x)
    dst[i] = (bf16)src[i];
}

// fp32 [R][Cc] -> bf16 transposed [Cc][R]
__global__ __launch_bounds__(256) void transpose_bf16_kernel(
    const float* __restrict__ src, bf16* __restrict__ dst, int R, int Cc) {
  int i = blockIdx.x * blockDim.x + threadIdx.x;
  if (i < R * Cc) {
    int r = i / Cc, c = i - r * Cc;
    dst[(size_t)c * R + r] = (bf16)src[i];
  }
}

// ---------------------------------------------------------------------------
// Register-blocked WMMA GEMM:  Cout[m][n] = sum_k A[g(m)][k]*Bt[n][k] + bias[n]
// One wave computes a 16 x 64 strip (4 accumulators). Direct global fragment
// loads (operands L2-resident: 192 MB L2 >> 48 MB A + 5 MB B). True ping-pong
// double buffering (k-loop unrolled by 2, two named buffer sets) so the
// rotation costs zero v_mov's and loads stay >1 iteration ahead of the
// matrix pipe. Requires Kdim % 64 == 0 (both call sites use 768).
// Optional row gather fuses the `inverse` permutation into the proj GEMM.
// ---------------------------------------------------------------------------
__global__ __launch_bounds__(256) void gemm_bf16_kernel(
    const bf16* __restrict__ A, int lda,
    const bf16* __restrict__ Bt, int ldb,
    const float* __restrict__ bias,
    float* __restrict__ Cout, int ldc,
    int Kdim, int Mstrips, const int* __restrict__ gather) {
  int wid  = (blockIdx.x * blockDim.x + threadIdx.x) >> 5;
  int lane = threadIdx.x & 31;
  int mstrip = wid % Mstrips;
  int nstrip = wid / Mstrips;      // 64-wide column strip
  int m0 = mstrip * 16;
  int n0 = nstrip * 64;
  int ml = lane & 15, sel = lane >> 4;

  int arow = m0 + ml;
  if (gather) arow = gather[arow];
  const bf16* Abase = A  + (size_t)arow * lda + sel * 8;
  const bf16* Bbase = Bt + (size_t)(n0 + ml) * ldb + sel * 8;

  v8f acc[4] = {};
  bf16x16 afA, afB, bA[4], bB[4];

  auto load_set = [&](bf16x16& af, bf16x16 (&bf)[4], int koff) {
    af = frag_ld(Abase + koff, Abase + koff + 16);
#pragma unroll
    for (int c = 0; c < 4; ++c) {
      const bf16* bp = Bbase + (size_t)(c * 16) * ldb + koff;
      bf[c] = frag_ld(bp, bp + 16);
    }
  };
  auto mm = [&](bf16x16& af, bf16x16 (&bf)[4]) {
#pragma unroll
    for (int c = 0; c < 4; ++c) acc[c] = wmma_bf16(af, bf[c], acc[c]);
  };

  load_set(afA, bA, 0);
  int k0 = 0;
  for (; k0 + 64 < Kdim; k0 += 64) {
    load_set(afB, bB, k0 + 32);   // in flight while A-set multiplies
    mm(afA, bA);
    load_set(afA, bA, k0 + 64);   // in flight while B-set multiplies
    mm(afB, bB);
  }
  load_set(afB, bB, Kdim - 32);   // final odd step
  mm(afA, bA);
  mm(afB, bB);

  int col = ml, rb = sel * 8;
#pragma unroll
  for (int r = 0; r < 8; ++r) {
    float* o = Cout + (size_t)(m0 + rb + r) * ldc + n0;
#pragma unroll
    for (int c = 0; c < 4; ++c)
      o[c * 16 + col] = acc[c][r] + bias[n0 + c * 16 + col];
  }
}

// ---------------------------------------------------------------------------
// Gather(order) + RoPE3D + layout transform.
// Writes Q (scaled by HD^-0.5) and K as [P][H][256][96] bf16 and V transposed
// [P][H][96][256] bf16 (B-frag friendly for the P·V matmul).
// ---------------------------------------------------------------------------
__global__ __launch_bounds__(256) void rope_kernel(
    const float* __restrict__ qkv, const int* __restrict__ order,
    const int* __restrict__ grid_coord,
    bf16* __restrict__ Qo, bf16* __restrict__ Ko, bf16* __restrict__ Vt) {
  int i = blockIdx.x;            // sorted index 0..N-1
  int src = order[i];
  const float* row = qkv + (size_t)src * C3;
  float pos0 = (float)grid_coord[src * 3 + 0];
  float pos1 = (float)grid_coord[src * 3 + 1];
  float pos2 = (float)grid_coord[src * 3 + 2];
  int p = i >> 8, ki = i & 255;
  const float scale = 0.10206207261596575f;   // 96^-0.5

  for (int e = threadIdx.x; e < CCH; e += 256) {
    int h = e / HD, d = e - h * HD;
    int a = d >> 5, u = d & 31, j = u & 15;
    bool hi = u >= 16;
    float pa = (a == 0) ? pos0 : (a == 1 ? pos1 : pos2);
    float invf = __powf(100.0f, -(float)j * (1.0f / 16.0f));
    float ang = pa * invf;
    float cs = __cosf(ang), sn = __sinf(ang);
    int base = h * HD + a * 32;

    float x1q = row[base + j],        x2q = row[base + 16 + j];
    float x1k = row[CCH + base + j],  x2k = row[CCH + base + 16 + j];
    float vq = hi ? (x1q * sn + x2q * cs) : (x1q * cs - x2q * sn);
    float vk = hi ? (x1k * sn + x2k * cs) : (x1k * cs - x2k * sn);
    float vv = row[2 * CCH + e];

    size_t qkbase = ((size_t)(p * HEADS + h) * WIN + ki) * HD + d;
    Qo[qkbase] = (bf16)(vq * scale);
    Ko[qkbase] = (bf16)vk;
    Vt[((size_t)(p * HEADS + h) * HD + d) * WIN + ki] = (bf16)vv;
  }
}

// ---------------------------------------------------------------------------
// Windowed attention. Grid = P*H*2 blocks, 8 waves/block; each wave owns a
// 16-query-row strip.
//   * K (256x96) and V^T (96x256) tiles shared by all 8 waves: staged ONCE
//     per block into LDS with global_load_async_to_lds_b128. All K loads are
//     issued before all V loads, so s_wait_asynccnt 12 releases the QK^T
//     phase while the V fill is still streaming (hidden under WMMA+softmax);
//     only the P·V phase waits for asynccnt==0.
//   * Full 16x256 logit strip in registers (16 v8f accumulators); softmax via
//     cross-lane shuffles inside the 16-lane half that owns each row (C/D
//     layout); P converted C-layout -> A-frag layout via 1 KB/wave LDS stage.
// Dynamic LDS: 48K (K) + 48K (V) + 8K (stage) = 104 KB; 2 blocks fit in the
// 320 KB WGP budget.
// ---------------------------------------------------------------------------
__global__ __launch_bounds__(256) void attn_kernel(
    const bf16* __restrict__ Q, const bf16* __restrict__ Kmat,
    const bf16* __restrict__ Vt, bf16* __restrict__ Out) {
  extern __shared__ __align__(16) char smem[];
  bf16* Ksh   = (bf16*)smem;              // [256][96]
  bf16* Vsh   = Ksh + WIN * HD;           // [96][256]
  bf16* stg   = Vsh + HD * WIN;           // 8 waves x 512

  int wave = threadIdx.x >> 5;
  int lane = threadIdx.x & 31;
  int half = blockIdx.x & 1;
  int ph   = blockIdx.x >> 1;          // p*8 + h
  int p = ph >> 3, h = ph & 7;

  const bf16* qb = Q    + (size_t)ph * WIN * HD;
  const bf16* kb = Kmat + (size_t)ph * WIN * HD;
  const bf16* vb = Vt   + (size_t)ph * HD * WIN;

  // ---- async stage of K then V tiles into LDS (once per block) ----
  unsigned k_lds0 = (unsigned)(uintptr_t)Ksh;   // low 32 bits = LDS offset
  unsigned v_lds0 = (unsigned)(uintptr_t)Vsh;
#pragma unroll
  for (int t = 0; t < (WIN * HD) / (256 * 8); ++t) {        // 12 K loads
    unsigned off = (threadIdx.x + t * 256) * 16;
    async_ld_b128(k_lds0 + off, off, kb);
  }
#pragma unroll
  for (int t = 0; t < (WIN * HD) / (256 * 8); ++t) {        // 12 V loads
    unsigned off = (threadIdx.x + t * 256) * 16;
    async_ld_b128(v_lds0 + off, off, vb);
  }

  int q0 = half * 128 + wave * 16;
  int ml = lane & 15, sel = lane >> 4, rb = sel * 8;

  // Q fragments (global; overlaps with the async fill)
  const bf16* qrow = qb + (size_t)(q0 + ml) * HD + sel * 8;
  bf16x16 qf0 = frag_ld(qrow +  0, qrow + 16);
  bf16x16 qf1 = frag_ld(qrow + 32, qrow + 48);
  bf16x16 qf2 = frag_ld(qrow + 64, qrow + 80);

  wait_async12();          // K tile resident (V still streaming)
  __syncthreads();

  // S = Q · K^T  (16 rows x 256 keys, in registers; K frags from LDS)
  v8f S[16];
#pragma unroll
  for (int c = 0; c < 16; ++c) {
    const bf16* krow = Ksh + (c * 16 + ml) * HD + sel * 8;
    v8f acc = {};
    acc = wmma_bf16(qf0, frag_ld(krow +  0, krow + 16), acc);
    acc = wmma_bf16(qf1, frag_ld(krow + 32, krow + 48), acc);
    acc = wmma_bf16(qf2, frag_ld(krow + 64, krow + 80), acc);
    S[c] = acc;
  }

  // Row softmax. Row (rb+r) lives across 16 lanes of this lane's half, so
  // shuffle masks 1,2,4,8 complete the reduction without crossing halves.
  float rinv[8], rmax[8];
#pragma unroll
  for (int r = 0; r < 8; ++r) {
    float m = -1e30f;
#pragma unroll
    for (int c = 0; c < 16; ++c) m = fmaxf(m, S[c][r]);
#pragma unroll
    for (int o = 1; o < 16; o <<= 1) m = fmaxf(m, __shfl_xor(m, o, 32));
    rmax[r] = m;
  }
#pragma unroll
  for (int r = 0; r < 8; ++r) {
    float s = 0.0f;
#pragma unroll
    for (int c = 0; c < 16; ++c) {
      float e = __expf(S[c][r] - rmax[r]);
      S[c][r] = e;
      s += e;
    }
#pragma unroll
    for (int o = 1; o < 16; o <<= 1) s += __shfl_xor(s, o, 32);
    rinv[r] = 1.0f / s;
  }

  wait_async0();           // V tile resident (fill hidden under QK^T+softmax)
  __syncthreads();

  // O = P · V  (contraction over 256 keys in 8 steps of 32; V frags from LDS)
  v8f O[6] = {};
  bf16* st = stg + wave * 512;
  for (int s = 0; s < 8; ++s) {
    // stage P tiles c = 2s, 2s+1 as bf16 [16 rows][32 keys] (C -> A layout)
#pragma unroll
    for (int cc = 0; cc < 2; ++cc) {
      int c = 2 * s + cc;
#pragma unroll
      for (int r = 0; r < 8; ++r)
        st[(rb + r) * 32 + cc * 16 + ml] = (bf16)(S[c][r] * rinv[r]);
    }
    const bf16* ar = st + ml * 32 + sel * 8;   // wave-private region:
    bf16x16 pf = frag_ld(ar, ar + 16);         // per-wave DS ops are in order
#pragma unroll
    for (int c6 = 0; c6 < 6; ++c6) {
      const bf16* vr = Vsh + (c6 * 16 + ml) * WIN + s * 32 + sel * 8;
      O[c6] = wmma_bf16(pf, frag_ld(vr, vr + 16), O[c6]);
    }
  }

  // Store to attn-out (sorted order) as bf16 [N][768]
#pragma unroll
  for (int r = 0; r < 8; ++r) {
    bf16* orow = Out + (size_t)(p * WIN + q0 + rb + r) * CCH + h * HD;
#pragma unroll
    for (int c6 = 0; c6 < 6; ++c6)
      orow[c6 * 16 + ml] = (bf16)O[c6][r];
  }
}

// ---------------------------------------------------------------------------
// Host-side orchestration
// ---------------------------------------------------------------------------
static inline size_t align256(size_t x) { return (x + 255) & ~(size_t)255; }

extern "C" void kernel_launch(void* const* d_in, const int* in_sizes, int n_in,
                              void* d_out, int out_size, void* d_ws, size_t ws_size,
                              hipStream_t stream) {
  const float* feat       = (const float*)d_in[0];
  const int*   grid_coord = (const int*)  d_in[1];
  const int*   order      = (const int*)  d_in[2];
  const int*   inverse    = (const int*)  d_in[3];
  const float* w_qkv      = (const float*)d_in[4];
  const float* b_qkv      = (const float*)d_in[5];
  const float* w_proj     = (const float*)d_in[6];
  const float* b_proj     = (const float*)d_in[7];
  float* out = (float*)d_out;

  char* w = (char*)d_ws;
  bf16* feat_bf = (bf16*)w;              w += align256((size_t)NPTS * CCH * 2);
  bf16* wqkvT   = (bf16*)w;              w += align256((size_t)C3 * CCH * 2);
  bf16* wprojT  = (bf16*)w;              w += align256((size_t)CCH * CCH * 2);
  float* qkv    = (float*)w;             w += align256((size_t)NPTS * C3 * 4);
  bf16* Qo      = (bf16*)w;              w += align256((size_t)NPTS * CCH * 2);
  bf16* Ko      = (bf16*)w;              w += align256((size_t)NPTS * CCH * 2);
  bf16* Vt      = (bf16*)w;              w += align256((size_t)NPTS * CCH * 2);
  bf16* attnout = (bf16*)w;              w += align256((size_t)NPTS * CCH * 2);

  // 1) precision conversion (+ weight transposes so B-frags are contiguous)
  cvt_bf16_kernel<<<4096, 256, 0, stream>>>(feat, feat_bf, NPTS * CCH);
  transpose_bf16_kernel<<<(CCH * C3 + 255) / 256, 256, 0, stream>>>(
      w_qkv, wqkvT, CCH, C3);
  transpose_bf16_kernel<<<(CCH * CCH + 255) / 256, 256, 0, stream>>>(
      w_proj, wprojT, CCH, CCH);

  // 2) qkv = feat @ w_qkv + b_qkv   (M=32768, N=2304, K=768)
  {
    int mstrips = NPTS / 16;             // 2048
    int nstrips = C3 / 64;               // 36
    int blocks  = mstrips * nstrips / 8; // 9216 blocks of 8 waves
    gemm_bf16_kernel<<<blocks, 256, 0, stream>>>(
        feat_bf, CCH, wqkvT, CCH, b_qkv, qkv, C3, CCH, mstrips, nullptr);
  }

  // 3) gather(order) + RoPE + attention-friendly layouts
  rope_kernel<<<NPTS, 256, 0, stream>>>(qkv, order, grid_coord, Qo, Ko, Vt);

  // 4) windowed attention (P*H*2 blocks, 104 KB dynamic LDS)
  {
    size_t shmem = (size_t)(WIN * HD + HD * WIN + 8 * 512) * sizeof(bf16);
    attn_kernel<<<PWIN * HEADS * 2, 256, shmem, stream>>>(Qo, Ko, Vt, attnout);
  }

  // 5) out = attnout[inverse] @ w_proj + b_proj  (M=32768, N=768, K=768)
  {
    int mstrips = NPTS / 16;             // 2048
    int nstrips = CCH / 64;              // 12
    int blocks  = mstrips * nstrips / 8; // 3072
    gemm_bf16_kernel<<<blocks, 256, 0, stream>>>(
        attnout, CCH, wprojT, CCH, b_proj, out, CCH, CCH, mstrips, inverse);
  }
}
